// MiniGPT_7576322310413
// MI455X (gfx1250) — compile-verified, matrix operand order checked by reading, not run on hardware
//
#include <hip/hip_runtime.h>
#include <hip/hip_bf16.h>
#include <math.h>

// MiniGPT fwd on gfx1250: all GEMMs via v_wmma_f32_16x16x32_f16 (wave32),
// attention (T=64, HD=20) in VALU f32, K padded 80->96 with zero columns.

#define BB   512
#define TT   64
#define DD   80
#define DP   96      // K padded to multiple of 32
#define HH   4
#define HDIM 20
#define LL   3
#define VV   8192
#define FF   320
#define MTOK (BB * TT)   // 32768 rows, multiple of 128

typedef _Float16 v16h __attribute__((ext_vector_type(16)));
typedef _Float16 v8h  __attribute__((ext_vector_type(8)));
typedef float    v8f  __attribute__((ext_vector_type(8)));

union HFrag { v16h v; v8h h[2]; };

// Per-lane 16x32 (or 32x16 mirrored) f16 fragment load, ISA layout:
// lanes 0-15: K = 0..7 then 16..23 ; lanes 16-31: K = 8..15 then 24..31.
// p = row base + k0 (16B aligned), kb = (lane>=16) ? 8 : 0.
__device__ __forceinline__ v16h load_frag16(const _Float16* __restrict__ p, int kb) {
  HFrag f;
  f.h[0] = *(const v8h*)(p + kb);
  f.h[1] = *(const v8h*)(p + 16 + kb);
  return f.v;
}

__device__ __forceinline__ float gelu_exact(float x) {
  return 0.5f * x * (1.0f + erff(x * 0.70710678118654752f));
}

// MODE 0: out f32 = acc + bias
// MODE 1: out f32 = out + acc + bias   (residual accumulate, in-place on h)
// MODE 2: out f16 = gelu(acc + bias)
template<int MODE>
__global__ void __launch_bounds__(128)
gemm_wmma(const _Float16* __restrict__ A, int lda,
          const _Float16* __restrict__ Bm, int ldb,
          const float* __restrict__ bias,
          void* __restrict__ outp, int ldc, int N, int Kp)
{
  const int lane = threadIdx.x & 31;
  const int wave = threadIdx.x >> 5;
  const int l16  = lane & 15;
  const int hi   = lane >> 4;
  const int kb   = hi * 8;
  const int mbase = blockIdx.y * 128 + wave * 32;
  const int nbase = blockIdx.x * 32;
  const int n0 = nbase + l16;
  const int n1 = nbase + 16 + l16;
  const bool n0ok = (n0 < N), n1ok = (n1 < N);

  const _Float16* a0p = A + (size_t)(mbase + l16) * lda;
  const _Float16* a1p = a0p + (size_t)16 * lda;
  // Clamp OOB columns to row 0: loaded garbage only feeds accumulators that
  // are never stored (guarded below); keeps loads unpredicated & EXEC uniform.
  const _Float16* b0p = Bm + (size_t)(n0ok ? n0 : 0) * ldb;
  const _Float16* b1p = Bm + (size_t)(n1ok ? n1 : 0) * ldb;

  v8f c00 = {}, c01 = {}, c10 = {}, c11 = {};
  for (int k0 = 0; k0 < Kp; k0 += 32) {
    v16h a0 = load_frag16(a0p + k0, kb);
    v16h a1 = load_frag16(a1p + k0, kb);
    v16h b0 = load_frag16(b0p + k0, kb);
    v16h b1 = load_frag16(b1p + k0, kb);
    c00 = __builtin_amdgcn_wmma_f32_16x16x32_f16(false, a0, false, b0, (short)0, c00, false, false);
    c01 = __builtin_amdgcn_wmma_f32_16x16x32_f16(false, a0, false, b1, (short)0, c01, false, false);
    c10 = __builtin_amdgcn_wmma_f32_16x16x32_f16(false, a1, false, b0, (short)0, c10, false, false);
    c11 = __builtin_amdgcn_wmma_f32_16x16x32_f16(false, a1, false, b1, (short)0, c11, false, false);
  }

  float bn0 = 0.f, bn1 = 0.f;
  if (bias) { if (n0ok) bn0 = bias[n0]; if (n1ok) bn1 = bias[n1]; }
  const int r0 = mbase + hi * 8;   // C/D layout: VGPR r -> M = r0 + r (tile0), +16 (tile1)

  if (MODE == 2) {
    _Float16* out16 = (_Float16*)outp;
#pragma unroll
    for (int r = 0; r < 8; ++r) {
      size_t i0 = (size_t)(r0 + r) * ldc;
      size_t i1 = (size_t)(r0 + 16 + r) * ldc;
      if (n0ok) { out16[i0 + n0] = (_Float16)gelu_exact(c00[r] + bn0);
                  out16[i1 + n0] = (_Float16)gelu_exact(c10[r] + bn0); }
      if (n1ok) { out16[i0 + n1] = (_Float16)gelu_exact(c01[r] + bn1);
                  out16[i1 + n1] = (_Float16)gelu_exact(c11[r] + bn1); }
    }
  } else {
    float* outf = (float*)outp;
#pragma unroll
    for (int r = 0; r < 8; ++r) {
      size_t i0 = (size_t)(r0 + r) * ldc;
      size_t i1 = (size_t)(r0 + 16 + r) * ldc;
      if (n0ok) {
        float u = c00[r] + bn0, w = c10[r] + bn0;
        if (MODE == 1) { u += outf[i0 + n0]; w += outf[i1 + n0]; }
        outf[i0 + n0] = u; outf[i1 + n0] = w;
      }
      if (n1ok) {
        float u = c01[r] + bn1, w = c11[r] + bn1;
        if (MODE == 1) { u += outf[i0 + n1]; w += outf[i1 + n1]; }
        outf[i0 + n1] = u; outf[i1 + n1] = w;
      }
    }
  }
}

// One wave32 per row: LN over D=80, emit f16 row of width DP=96 (cols 80..95 zero).
__global__ void __launch_bounds__(128)
ln_to_f16(const float* __restrict__ h, const float* __restrict__ gamma,
          const float* __restrict__ beta, _Float16* __restrict__ out)
{
  const int lane = threadIdx.x & 31;
  const int row  = blockIdx.x * 4 + (threadIdx.x >> 5);
  const float* hr = h + (size_t)row * DD;
  float x0 = hr[lane];
  float x1 = hr[lane + 32];
  float x2 = (lane < 16) ? hr[lane + 64] : 0.f;
  float s = x0 + x1 + x2;
#pragma unroll
  for (int off = 16; off >= 1; off >>= 1) s += __shfl_xor(s, off, 32);
  float mean = s * (1.0f / DD);
  float d0 = x0 - mean, d1 = x1 - mean;
  float d2 = (lane < 16) ? (x2 - mean) : 0.f;
  float v = d0 * d0 + d1 * d1 + d2 * d2;
#pragma unroll
  for (int off = 16; off >= 1; off >>= 1) v += __shfl_xor(v, off, 32);
  float inv = rsqrtf(v * (1.0f / DD) + 1e-5f);
  _Float16* orow = out + (size_t)row * DP;
  orow[lane]      = (_Float16)(d0 * inv * gamma[lane] + beta[lane]);
  orow[lane + 32] = (_Float16)(d1 * inv * gamma[lane + 32] + beta[lane + 32]);
  float y2 = (lane < 16) ? (d2 * inv * gamma[lane + 64] + beta[lane + 64]) : 0.f;
  orow[lane + 64] = (_Float16)y2;   // lanes>=16 write zeros to cols 80..95
}

__global__ void embed_kernel(const int* __restrict__ x, const float* __restrict__ tok,
                             const float* __restrict__ pos, float* __restrict__ h)
{
  int i = blockIdx.x * blockDim.x + threadIdx.x;
  if (i >= MTOK * DD) return;
  int m = i / DD, d = i - m * DD;
  int t = m & (TT - 1);
  h[i] = tok[(size_t)x[m] * DD + d] + pos[t * DD + d];
}

// Causal softmax attention, one block per (batch, head); thread = query row.
__global__ void __launch_bounds__(64)
attn_kernel(const float* __restrict__ qkv, _Float16* __restrict__ attn16)
{
  __shared__ float sc[TT * TT];   // 16 KB
  const int b = blockIdx.x >> 2;
  const int head = blockIdx.x & 3;
  const int t = threadIdx.x;
  const size_t mrow = (size_t)(b * TT + t);
  const float* qrow = qkv + mrow * 240 + head * HDIM;
  float q[HDIM];
#pragma unroll
  for (int d = 0; d < HDIM; ++d) q[d] = qrow[d];
  const float scale = 0.22360679774997896f;  // HD^-0.5
  float* myrow = sc + t * TT;
  float mx = -1e30f;
  for (int tk = 0; tk <= t; ++tk) {
    const float* krow = qkv + (size_t)(b * TT + tk) * 240 + 80 + head * HDIM;
    float s = 0.f;
#pragma unroll
    for (int d = 0; d < HDIM; ++d) s += q[d] * krow[d];
    s *= scale;
    myrow[tk] = s;
    mx = fmaxf(mx, s);
  }
  float sum = 0.f;
  for (int tk = 0; tk <= t; ++tk) {
    float w = expf(myrow[tk] - mx);
    myrow[tk] = w;
    sum += w;
  }
  float acc[HDIM];
#pragma unroll
  for (int d = 0; d < HDIM; ++d) acc[d] = 0.f;
  for (int tk = 0; tk <= t; ++tk) {
    float w = myrow[tk];
    const float* vrow = qkv + (size_t)(b * TT + tk) * 240 + 160 + head * HDIM;
#pragma unroll
    for (int d = 0; d < HDIM; ++d) acc[d] += w * vrow[d];
  }
  float invs = 1.0f / sum;
  _Float16* orow = attn16 + mrow * DP + head * HDIM;
#pragma unroll
  for (int d = 0; d < HDIM; ++d) orow[d] = (_Float16)(acc[d] * invs);
  if (head == 0) {                // zero pad columns 80..95
    _Float16* prow = attn16 + mrow * DP + 80;
#pragma unroll
    for (int d = 0; d < 16; ++d) prow[d] = (_Float16)0.f;
  }
}

// f32 (rows,K) -> f16 (rows,Kp) with zero-padded K tail.
__global__ void convert_pad_kernel(const float* __restrict__ src, _Float16* __restrict__ dst,
                                   int rows, int K, int Kp)
{
  int i = blockIdx.x * blockDim.x + threadIdx.x;
  if (i >= rows * Kp) return;
  int r = i / Kp, c = i - r * Kp;
  dst[i] = (c < K) ? (_Float16)src[(size_t)r * K + c] : (_Float16)0.f;
}

extern "C" void kernel_launch(void* const* d_in, const int* in_sizes, int n_in,
                              void* d_out, int out_size, void* d_ws, size_t ws_size,
                              hipStream_t stream)
{
  (void)in_sizes; (void)n_in; (void)out_size; (void)ws_size;
  const int*   x      = (const int*)  d_in[0];
  const float* tok    = (const float*)d_in[1];
  const float* pos    = (const float*)d_in[2];
  const float* ln1_s  = (const float*)d_in[3];
  const float* ln1_b  = (const float*)d_in[4];
  const float* ln2_s  = (const float*)d_in[5];
  const float* ln2_b  = (const float*)d_in[6];
  const float* qkv_w  = (const float*)d_in[7];
  const float* qkv_b  = (const float*)d_in[8];
  const float* out_w  = (const float*)d_in[9];
  const float* out_b  = (const float*)d_in[10];
  const float* ffn_w1 = (const float*)d_in[11];
  const float* ffn_b1 = (const float*)d_in[12];
  const float* ffn_w2 = (const float*)d_in[13];
  const float* ffn_b2 = (const float*)d_in[14];
  const float* lnf_s  = (const float*)d_in[15];
  const float* lnf_b  = (const float*)d_in[16];

  char* ws = (char*)d_ws;
  size_t off = 0;
  auto alloc = [&](size_t bytes) -> void* {
    void* p = ws + off;
    off = (off + bytes + 255) & ~(size_t)255;
    return p;
  };
  float*    h      = (float*)   alloc((size_t)MTOK * DD * 4);   // residual stream
  _Float16* a16    = (_Float16*)alloc((size_t)MTOK * DP * 2);   // LN activations (padded)
  float*    qkvf   = (float*)   alloc((size_t)MTOK * 240 * 4);  // qkv projections
  _Float16* attn16 = (_Float16*)alloc((size_t)MTOK * DP * 2);   // attention out (padded)
  _Float16* g16    = (_Float16*)alloc((size_t)MTOK * FF * 2);   // gelu(FFN1)
  _Float16* wqkv16 = (_Float16*)alloc((size_t)LL * 240 * DP * 2);
  _Float16* wout16 = (_Float16*)alloc((size_t)LL * DD * DP * 2);
  _Float16* wf116  = (_Float16*)alloc((size_t)LL * FF * DP * 2);
  _Float16* wf216  = (_Float16*)alloc((size_t)LL * DD * FF * 2);
  _Float16* tok16  = (_Float16*)alloc((size_t)VV * DP * 2);

  auto cvt = [&](const float* s, _Float16* d, int rows, int K, int Kp) {
    int tot = rows * Kp;
    convert_pad_kernel<<<(tot + 255) / 256, 256, 0, stream>>>(s, d, rows, K, Kp);
  };
  cvt(qkv_w,  wqkv16, LL * 240, DD, DP);
  cvt(out_w,  wout16, LL * DD,  DD, DP);
  cvt(ffn_w1, wf116,  LL * FF,  DD, DP);
  cvt(ffn_w2, wf216,  LL * DD,  FF, FF);
  cvt(tok,    tok16,  VV,       DD, DP);

  embed_kernel<<<(MTOK * DD + 255) / 256, 256, 0, stream>>>(x, tok, pos, h);

  const dim3 blk(128);
  const int mblocks = MTOK / 128;
  for (int l = 0; l < LL; ++l) {
    ln_to_f16<<<MTOK / 4, 128, 0, stream>>>(h, ln1_s + l * DD, ln1_b + l * DD, a16);
    gemm_wmma<0><<<dim3(8, mblocks), blk, 0, stream>>>(
        a16, DP, wqkv16 + (size_t)l * 240 * DP, DP, qkv_b + l * 240, qkvf, 240, 240, DP);
    attn_kernel<<<BB * HH, 64, 0, stream>>>(qkvf, attn16);
    gemm_wmma<1><<<dim3(3, mblocks), blk, 0, stream>>>(
        attn16, DP, wout16 + (size_t)l * DD * DP, DP, out_b + l * DD, h, DD, DD, DP);
    ln_to_f16<<<MTOK / 4, 128, 0, stream>>>(h, ln2_s + l * DD, ln2_b + l * DD, a16);
    gemm_wmma<2><<<dim3(10, mblocks), blk, 0, stream>>>(
        a16, DP, wf116 + (size_t)l * FF * DP, DP, ffn_b1 + l * FF, g16, FF, FF, DP);
    gemm_wmma<1><<<dim3(3, mblocks), blk, 0, stream>>>(
        g16, FF, wf216 + (size_t)l * DD * FF, FF, ffn_b2 + l * DD, h, DD, DD, FF);
  }
  ln_to_f16<<<MTOK / 4, 128, 0, stream>>>(h, lnf_s, lnf_b, a16);
  gemm_wmma<0><<<dim3(VV / 32, mblocks), blk, 0, stream>>>(
      a16, DP, tok16, DP, nullptr, d_out, VV, VV, DP);
}
